// GNNModel_37632503447782
// MI455X (gfx1250) — matrix operand order repaired and checked by reading, not compiled, not run on hardware
//
#include <hip/hip_runtime.h>
#include <math.h>

// GCN 2-layer forward for MI455X (gfx1250, wave32).
// Strategy: tiny GEMMs via v_wmma_f32_16x16x32_f16 (fp32 accum); the dominant
// cost (edge scatter) uses hardware global_atomic_add_f32 into L2-resident
// workspace buffers (~72MB < 192MB L2).

#define F_IN   128
#define F_HID  64
#define F_OUT  47
#define F_OUT_P 48   // padded to 3 WMMA tiles

typedef __attribute__((ext_vector_type(16))) _Float16 v16h;
typedef __attribute__((ext_vector_type(8)))  float    v8f;

__device__ __forceinline__ void atom_add_f32(float* p, float v) {
  // non-returning HW float atomic (global_atomic_add_f32, no pre-op return)
  unsafeAtomicAdd(p, v);
}

// ---------------- degree pipeline ----------------

__global__ void k_deg_init(float* deg, int n) {
  int i = blockIdx.x * blockDim.x + threadIdx.x;
  if (i < n) deg[i] = 1.0f;  // self-loop
}

__global__ void k_deg_count(const int* __restrict__ dst, float* deg, int e) {
  int i = blockIdx.x * blockDim.x + threadIdx.x;
  if (i < e) atom_add_f32(&deg[dst[i]], 1.0f);
}

__global__ void k_deg_inv(const float* __restrict__ deg,
                          float* __restrict__ dinv, float* __restrict__ invd, int n) {
  int i = blockIdx.x * blockDim.x + threadIdx.x;
  if (i < n) {
    float d = deg[i];
    dinv[i] = rsqrtf(d);
    invd[i] = 1.0f / d;
  }
}

// ---------------- WMMA GEMM 1: xw1 = x(N x 128) @ W1(128 x 64) ----------------
// Block = 128 threads = 4 waves. Block computes rows [16*bid,16*bid+16),
// wave w computes output columns [16w, 16w+16). K swept in 4 chunks of 32.

__global__ void k_gemm1(const float* __restrict__ x, const float* __restrict__ W,
                        float* __restrict__ out, int n) {
  const int lane = threadIdx.x & 31;
  const int wave = threadIdx.x >> 5;
  const int row0 = blockIdx.x * 16;
  const int col0 = wave * 16;
  const int r  = lane & 15;   // A-row / B-col handled by this lane
  const int hi = lane >> 4;   // lane half selects K sub-block (ISA 7.12.2)
  int arow = row0 + r; arow = arow < n ? arow : (n - 1);   // branchless clamp
  const int kbA = hi * 16;    // A: lanes16-31 hold K=16..31
  const int kbB = hi * 16;    // B: lanes16-31 hold K=16..31

  v8f acc = {};
  for (int k0 = 0; k0 < F_IN; k0 += 32) {
    v16h a, b;
    const float* xr = x + (size_t)arow * F_IN + k0;
#pragma unroll
    for (int j = 0; j < 8; ++j) {
      a[j]     = (_Float16)xr[kbA + j];
      a[j + 8] = (_Float16)xr[kbA + 8 + j];
    }
    const float* wc = W + (size_t)(k0 + kbB) * F_HID + (col0 + r);
#pragma unroll
    for (int j = 0; j < 16; ++j) b[j] = (_Float16)wc[(size_t)j * F_HID];
    acc = __builtin_amdgcn_wmma_f32_16x16x32_f16(false, a, false, b,
                                                 (short)0, acc, false, false);
  }
  // C/D layout: lane -> N, VGPR v -> M = v + 8*hi
  const int oc = col0 + r;
#pragma unroll
  for (int v = 0; v < 8; ++v) {
    int orow = row0 + hi * 8 + v;
    if (orow < n) out[(size_t)orow * F_HID + oc] = acc[v];
  }
}

// ---------------- WMMA GEMM 2: xw2 = h(N x 64) @ W2(64 x 47) -> stride 48 ----------------
// Block = 96 threads = 3 waves (48 output cols; col 47 padded with zero weights).

__global__ void k_gemm2(const float* __restrict__ h, const float* __restrict__ W,
                        float* __restrict__ out, int n) {
  const int lane = threadIdx.x & 31;
  const int wave = threadIdx.x >> 5;
  const int row0 = blockIdx.x * 16;
  const int col0 = wave * 16;
  const int r  = lane & 15;
  const int hi = lane >> 4;
  int arow = row0 + r; arow = arow < n ? arow : (n - 1);
  const int kbA = hi * 16;
  const int kbB = hi * 16;
  const int col = col0 + r;

  v8f acc = {};
  for (int k0 = 0; k0 < F_HID; k0 += 32) {
    v16h a, b;
    const float* hr = h + (size_t)arow * F_HID + k0;
#pragma unroll
    for (int j = 0; j < 8; ++j) {
      a[j]     = (_Float16)hr[kbA + j];
      a[j + 8] = (_Float16)hr[kbA + 8 + j];
    }
#pragma unroll
    for (int j = 0; j < 16; ++j) {
      float wv = (col < F_OUT) ? W[(size_t)(k0 + kbB + j) * F_OUT + col] : 0.0f;
      b[j] = (_Float16)wv;
    }
    acc = __builtin_amdgcn_wmma_f32_16x16x32_f16(false, a, false, b,
                                                 (short)0, acc, false, false);
  }
#pragma unroll
  for (int v = 0; v < 8; ++v) {
    int orow = row0 + hi * 8 + v;
    if (orow < n) out[(size_t)orow * F_OUT_P + col] = acc[v];
  }
}

// ---------------- self-loop init: agg[i][:] = xw[i][:] * (1/deg[i]) ----------------

__global__ void k_self_init(const float* __restrict__ xw, const float* __restrict__ invd,
                            float* __restrict__ agg, int n, int fdim) {
  int i = blockIdx.x * blockDim.x + threadIdx.x;      // one float4 group
  int ngroups = fdim >> 2;
  if (i >= n * ngroups) return;
  int node = i / ngroups, g = i - node * ngroups;
  float s = invd[node];
  const float4 v = *(const float4*)(xw + (size_t)node * fdim + 4 * g);
  float4 o; o.x = v.x * s; o.y = v.y * s; o.z = v.z * s; o.w = v.w * s;
  *(float4*)(agg + (size_t)node * fdim + 4 * g) = o;
}

// ---------------- edge scatter: agg[dst] += xw[src] * dinv[src]*dinv[dst] ----------------
// One thread per (edge, float4 group). Dominant kernel: random L2 gathers + fp atomics.

__global__ void k_scatter(const int* __restrict__ src, const int* __restrict__ dst,
                          const float* __restrict__ xw, const float* __restrict__ dinv,
                          float* __restrict__ agg, int e, int fdim) {
  int ngroups = fdim >> 2;
  int idx = blockIdx.x * blockDim.x + threadIdx.x;
  if (idx >= e * ngroups) return;
  int ed = idx / ngroups, g = idx - ed * ngroups;
  int s = src[ed], d = dst[ed];
  float nrm = dinv[s] * dinv[d];
  const float4 m = *(const float4*)(xw + (size_t)s * fdim + 4 * g);
  float* p = agg + (size_t)d * fdim + 4 * g;
  atom_add_f32(p + 0, m.x * nrm);
  atom_add_f32(p + 1, m.y * nrm);
  atom_add_f32(p + 2, m.z * nrm);
  atom_add_f32(p + 3, m.w * nrm);
}

// ---------------- bias + ReLU: h = max(agg1 + b1, 0) ----------------

__global__ void k_bias_relu(const float* __restrict__ agg, const float* __restrict__ b,
                            float* __restrict__ h, int n) {
  int i = blockIdx.x * blockDim.x + threadIdx.x;      // one float4 of F_HID
  const int ngroups = F_HID >> 2;
  if (i >= n * ngroups) return;
  int node = i / ngroups, g = i - node * ngroups;
  const float4 v = *(const float4*)(agg + (size_t)node * F_HID + 4 * g);
  const float4 bb = *(const float4*)(b + 4 * g);
  float4 o;
  o.x = fmaxf(v.x + bb.x, 0.0f); o.y = fmaxf(v.y + bb.y, 0.0f);
  o.z = fmaxf(v.z + bb.z, 0.0f); o.w = fmaxf(v.w + bb.w, 0.0f);
  *(float4*)(h + (size_t)node * F_HID + 4 * g) = o;
}

// ---------------- log-softmax over 47 classes, one wave32 per node ----------------

__global__ void k_logsoftmax(const float* __restrict__ agg, const float* __restrict__ b2,
                             float* __restrict__ out, int n) {
  int node = blockIdx.x * (blockDim.x >> 5) + (threadIdx.x >> 5);
  if (node >= n) return;
  int lane = threadIdx.x & 31;
  const float* a = agg + (size_t)node * F_OUT_P;
  float z0 = -INFINITY, z1 = -INFINITY;
  if (lane < F_OUT) z0 = a[lane] + b2[lane];
  int c1 = lane + 32;
  if (c1 < F_OUT) z1 = a[c1] + b2[c1];
  float m = fmaxf(z0, z1);
#pragma unroll
  for (int off = 16; off > 0; off >>= 1) m = fmaxf(m, __shfl_xor(m, off, 32));
  float s = 0.0f;
  if (lane < F_OUT) s += __expf(z0 - m);
  if (c1 < F_OUT)   s += __expf(z1 - m);
#pragma unroll
  for (int off = 16; off > 0; off >>= 1) s += __shfl_xor(s, off, 32);
  float lse = m + __logf(s);
  float* o = out + (size_t)node * F_OUT;
  if (lane < F_OUT) o[lane] = z0 - lse;
  if (c1 < F_OUT)   o[c1]   = z1 - lse;
}

// ---------------- launch ----------------

extern "C" void kernel_launch(void* const* d_in, const int* in_sizes, int n_in,
                              void* d_out, int out_size, void* d_ws, size_t ws_size,
                              hipStream_t stream) {
  const float* x  = (const float*)d_in[0];
  const int*   ei = (const int*)  d_in[1];
  const float* W1 = (const float*)d_in[2];
  const float* b1 = (const float*)d_in[3];
  const float* W2 = (const float*)d_in[4];
  const float* b2 = (const float*)d_in[5];
  float* out = (float*)d_out;

  const int n = in_sizes[0] / F_IN;
  const int e = in_sizes[1] / 2;
  const int* src = ei;
  const int* dst = ei + e;

  // workspace layout (reused buffers; total ~72MB, fits in 192MB L2)
  float* deg  = (float*)d_ws;                       // n
  float* dinv = deg  + n;                           // n
  float* invd = dinv + n;                           // n
  float* bufA = invd + n;                           // n*64 : xw1, then h
  float* bufB = bufA + (size_t)n * F_HID;           // n*64 : agg1, then xw2 (stride 48)
  float* bufC = bufB + (size_t)n * F_HID;           // n*48 : agg2

  const int T = 256;
  // 1) degree of A + I
  k_deg_init <<<(n + T - 1) / T, T, 0, stream>>>(deg, n);
  k_deg_count<<<(e + T - 1) / T, T, 0, stream>>>(dst, deg, e);
  k_deg_inv  <<<(n + T - 1) / T, T, 0, stream>>>(deg, dinv, invd, n);

  // 2) layer 1
  k_gemm1<<<(n + 15) / 16, 128, 0, stream>>>(x, W1, bufA, n);           // xw1 -> A
  {
    int work = n * (F_HID >> 2);
    k_self_init<<<(work + T - 1) / T, T, 0, stream>>>(bufA, invd, bufB, n, F_HID);
    int ework = e * (F_HID >> 2);
    k_scatter<<<(ework + T - 1) / T, T, 0, stream>>>(src, dst, bufA, dinv, bufB, e, F_HID);
    k_bias_relu<<<(work + T - 1) / T, T, 0, stream>>>(bufB, b1, bufA, n); // h -> A
  }

  // 3) layer 2 (stride-48 padded feature dim)
  k_gemm2<<<(n + 15) / 16, 96, 0, stream>>>(bufA, W2, bufB, n);         // xw2 -> B
  {
    int work = n * (F_OUT_P >> 2);
    k_self_init<<<(work + T - 1) / T, T, 0, stream>>>(bufB, invd, bufC, n, F_OUT_P);
    int ework = e * (F_OUT_P >> 2);
    k_scatter<<<(ework + T - 1) / T, T, 0, stream>>>(src, dst, bufB, dinv, bufC, e, F_OUT_P);
  }

  // 4) log-softmax (one wave per node, 8 nodes per 256-thread block)
  k_logsoftmax<<<(n + 7) / 8, 256, 0, stream>>>(bufC, b2, out, n);
}
